// MemoryUpdater_19499151524025
// MI455X (gfx1250) — matrix operand order repaired and checked by reading, not compiled, not run on hardware
//
#include <hip/hip_runtime.h>
#include <hip/hip_bf16.h>

typedef _Float16 v16h  __attribute__((ext_vector_type(16)));
typedef __fp16   fp16x2 __attribute__((ext_vector_type(2)));
typedef float    v8f   __attribute__((ext_vector_type(8)));
typedef float    f4raw __attribute__((ext_vector_type(4)));

#define D_S        128
#define HALF_MAT   49152   // 384*128 halves per matrix
#define HALF_TOT   98304   // two matrices
#define WS_BYTES   (HALF_TOT * 2)

// ---------------------------------------------------------------------------
// B-operand layout for v_wmma_f32_16x16x32_f16 (B is 32x16, K-major pairs):
//   lanes 0-15 : N = lane,    elements 0-7 hold K 0-7,  elements 8-15 hold K 16-23
//   lanes 16-31: N = lane-16, elements 0-7 hold K 8-15, elements 8-15 hold K 24-31
// Tile (jc, kc) of W^T covers output cols jc*16..+15, K = kc*32..+31.
// Each tile = 512 halves (1024B); lane's 8 VGPRs are 32 contiguous bytes.
// ---------------------------------------------------------------------------
__device__ __forceinline__ int perm_index(int j, int k) {
    int jc = j >> 4, ncol = j & 15;
    int kc = k >> 5, kr = k & 31;
    int lane = ((kr >> 3) & 1) ? (16 + ncol) : ncol;
    int s = (kr & 7) + ((kr >> 4) << 3);
    return ((jc << 2) + kc) * 512 + lane * 16 + s;
}

__device__ __forceinline__ v8f zero8() {
    v8f v = {0.f, 0.f, 0.f, 0.f, 0.f, 0.f, 0.f, 0.f};
    return v;
}

__device__ __forceinline__ v8f wmma16(v16h a, v16h b, v8f c) {
    // D = A(16x32 f16) x B(32x16 f16) + C(16x16 f32)
    return __builtin_amdgcn_wmma_f32_16x16x32_f16(false, a, false, b,
                                                  (short)0, c, false, false);
}

__device__ __forceinline__ float fast_exp(float x) {
#if __has_builtin(__builtin_amdgcn_exp2f)
    return __builtin_amdgcn_exp2f(x * 1.44269504088896340736f);
#else
    return __expf(x);
#endif
}
__device__ __forceinline__ float fast_rcp(float x) {
#if __has_builtin(__builtin_amdgcn_rcpf)
    return __builtin_amdgcn_rcpf(x);
#else
    return 1.0f / x;
#endif
}
__device__ __forceinline__ float fsig(float x)  { return fast_rcp(1.0f + fast_exp(-x)); }
__device__ __forceinline__ float ftanh(float x) {
    float t = fast_exp(2.0f * x);                 // saturates cleanly at +/-inf
    return 1.0f - 2.0f * fast_rcp(t + 1.0f);
}

// Load 8+8 consecutive f32 (first-half K at koff, second half at koff+16) into
// the per-lane A layout for 16x16x32 f16 WMMA using packed f32->f16 converts.
__device__ __forceinline__ v16h load_a16(const float* __restrict__ row, int koff) {
    const float4 a = *reinterpret_cast<const float4*>(row + koff);
    const float4 b = *reinterpret_cast<const float4*>(row + koff + 4);
    const float4 c = *reinterpret_cast<const float4*>(row + koff + 16);
    const float4 d = *reinterpret_cast<const float4*>(row + koff + 20);
    union { v16h v; fp16x2 h[8]; } u;
#if __has_builtin(__builtin_amdgcn_cvt_pkrtz)
    u.h[0] = __builtin_amdgcn_cvt_pkrtz(a.x, a.y);   // v_cvt_pk_rtz_f16_f32
    u.h[1] = __builtin_amdgcn_cvt_pkrtz(a.z, a.w);
    u.h[2] = __builtin_amdgcn_cvt_pkrtz(b.x, b.y);
    u.h[3] = __builtin_amdgcn_cvt_pkrtz(b.z, b.w);
    u.h[4] = __builtin_amdgcn_cvt_pkrtz(c.x, c.y);
    u.h[5] = __builtin_amdgcn_cvt_pkrtz(c.z, c.w);
    u.h[6] = __builtin_amdgcn_cvt_pkrtz(d.x, d.y);
    u.h[7] = __builtin_amdgcn_cvt_pkrtz(d.z, d.w);
#else
    u.v[0]  = (_Float16)a.x; u.v[1]  = (_Float16)a.y; u.v[2]  = (_Float16)a.z; u.v[3]  = (_Float16)a.w;
    u.v[4]  = (_Float16)b.x; u.v[5]  = (_Float16)b.y; u.v[6]  = (_Float16)b.z; u.v[7]  = (_Float16)b.w;
    u.v[8]  = (_Float16)c.x; u.v[9]  = (_Float16)c.y; u.v[10] = (_Float16)c.z; u.v[11] = (_Float16)c.w;
    u.v[12] = (_Float16)d.x; u.v[13] = (_Float16)d.y; u.v[14] = (_Float16)d.z; u.v[15] = (_Float16)d.w;
#endif
    return u.v;
}

// ---------------------------------------------------------------------------
__global__ void fill_ones_kernel(float* __restrict__ out, long long n) {
    long long i4 = ((long long)blockIdx.x * blockDim.x + threadIdx.x) * 4;
    if (i4 + 3 < n) {
        f4raw ones = {1.f, 1.f, 1.f, 1.f};
        __builtin_nontemporal_store(ones, reinterpret_cast<f4raw*>(out + i4));
    } else {
        for (long long t = i4; t < n; ++t) out[t] = 1.0f;
    }
}

__global__ void gru_pack_weights(const float* __restrict__ Wih,
                                 const float* __restrict__ Whh,
                                 _Float16* __restrict__ ws) {
    int i = blockIdx.x * 256 + threadIdx.x;
    if (i >= HALF_TOT) return;
    int m   = (i >= HALF_MAT);
    int rem = i - m * HALF_MAT;
    int j = rem >> 7, k = rem & 127;
    const float* W = m ? Whh : Wih;
    ws[m * HALF_MAT + perm_index(j, k)] = (_Float16)W[rem];
}

// ---------------------------------------------------------------------------
// 256 threads = 8 wave32; each wave computes 16 rows of h_new (16x128 tile).
// LDS: 192KB packed f16 weights + 2KB combined biases (<= 320KB WGP budget).
// r/z gates accumulate gi+gh into a single WMMA C tile each (4 accumulators
// instead of 6); all gather/scatter uses uniform-base + 32-bit offsets.
// ---------------------------------------------------------------------------
__global__ __launch_bounds__(256) void gru_wmma_kernel(
        const float* __restrict__ msgs, const float* __restrict__ S,
        const float* __restrict__ Wih,  const float* __restrict__ Whh,
        const float* __restrict__ bih,  const float* __restrict__ bhh,
        const long long* __restrict__ idx, const _Float16* __restrict__ wpk,
        float* __restrict__ out, int Mrows, int use_ws) {

    __shared__ __align__(32) _Float16 wlds[HALF_TOT];
    __shared__ float sbias[4 * D_S];   // s_r | s_z | b_in | b_hn
    const int tid = threadIdx.x;

    if (use_ws) {
        const float4* src = reinterpret_cast<const float4*>(wpk);
        float4*       dst = reinterpret_cast<float4*>(wlds);
        #pragma unroll 4
        for (int i = tid; i < WS_BYTES / 16; i += 256) dst[i] = src[i];
    } else {
        for (int i = tid; i < HALF_TOT; i += 256) {
            int m = (i >= HALF_MAT);
            int rem = i - m * HALF_MAT;
            int j = rem >> 7, k = rem & 127;
            const float* W = m ? Whh : Wih;
            wlds[m * HALF_MAT + perm_index(j, k)] = (_Float16)W[rem];
        }
    }
    if (tid < D_S) {
        sbias[tid]           = bih[tid]           + bhh[tid];
        sbias[D_S + tid]     = bih[D_S + tid]     + bhh[D_S + tid];
        sbias[2 * D_S + tid] = bih[2 * D_S + tid];
        sbias[3 * D_S + tid] = bhh[2 * D_S + tid];
    }
    __syncthreads();

    const int lane = tid & 31;
    const int wave = tid >> 5;
    const int rows_base = blockIdx.x * 128 + wave * 16;
    if (rows_base >= Mrows) return;          // wave-uniform: EXEC stays all-ones

    const int sel    = lane >> 4;            // 0: K 0-7/16-23, 1: K 8-15/24-31
    const int alocal = lane & 15;            // A row within tile

    const int arow = min(rows_base + alocal, Mrows - 1);
    const int hrow = (int)idx[arow];

    v16h ax[4], ah[4];
    {
        const float* xr = msgs + (size_t)arow * D_S;
        const float* hr = S    + (size_t)hrow * D_S;
        #pragma unroll
        for (int kc = 0; kc < 4; ++kc) {
            const int koff = kc * 32 + sel * 8;
            ax[kc] = load_a16(xr, koff);
            ah[kc] = load_a16(hr, koff);
        }
    }

    // C/D layout rows: VGPR e, lanes<16 -> M=e ; lanes>=16 -> M=e+8.
    // 32-bit element offsets (output is 25.6M floats -> fits easily).
    unsigned rbase[8];
    #pragma unroll
    for (int e = 0; e < 8; ++e) {
        int crow = min(rows_base + e + sel * 8, Mrows - 1);
        rbase[e] = ((unsigned)(int)idx[crow]) << 7;   // *128
    }

    for (int jc = 0; jc < 8; ++jc) {         // output column chunk of 16
        v8f accr = zero8(), accz = zero8();  // gi+gh fused for r and z
        v8f ain  = zero8(), ahn  = zero8();  // n gate parts kept separate

        const _Float16* lb = wlds + lane * 16;
        #pragma unroll
        for (int kc = 0; kc < 4; ++kc) {
            v16h b;
            b = *reinterpret_cast<const v16h*>(lb + ((jc)      * 4 + kc) * 512);
            accr = wmma16(ax[kc], b, accr);
            b = *reinterpret_cast<const v16h*>(lb + HALF_MAT + ((jc)      * 4 + kc) * 512);
            accr = wmma16(ah[kc], b, accr);
            b = *reinterpret_cast<const v16h*>(lb + ((8 + jc)  * 4 + kc) * 512);
            accz = wmma16(ax[kc], b, accz);
            b = *reinterpret_cast<const v16h*>(lb + HALF_MAT + ((8 + jc)  * 4 + kc) * 512);
            accz = wmma16(ah[kc], b, accz);
            b = *reinterpret_cast<const v16h*>(lb + ((16 + jc) * 4 + kc) * 512);
            ain  = wmma16(ax[kc], b, ain);
            b = *reinterpret_cast<const v16h*>(lb + HALF_MAT + ((16 + jc) * 4 + kc) * 512);
            ahn  = wmma16(ah[kc], b, ahn);
        }

        const unsigned col = (unsigned)(jc * 16 + alocal);   // C layout: N = lane & 15
        const float s_r  = sbias[col];
        const float s_z  = sbias[D_S + col];
        const float b_in = sbias[2 * D_S + col];
        const float b_hn = sbias[3 * D_S + col];

        #pragma unroll
        for (int e = 0; e < 8; ++e) {
            const float r = fsig(accr[e] + s_r);
            const float z = fsig(accz[e] + s_z);
            const float n = ftanh(ain[e] + b_in + r * (ahn[e] + b_hn));
            const unsigned off = rbase[e] + col;             // saddr + 32-bit voffset
            const float hv = S[off];
            out[off] = (1.0f - z) * n + z * hv;
        }
    }
}

// ---------------------------------------------------------------------------
extern "C" void kernel_launch(void* const* d_in, const int* in_sizes, int n_in,
                              void* d_out, int out_size, void* d_ws, size_t ws_size,
                              hipStream_t stream) {
    (void)n_in;
    const float*     msgs = (const float*)d_in[0];
    const float*     S    = (const float*)d_in[1];
    const float*     Wih  = (const float*)d_in[2];
    const float*     Whh  = (const float*)d_in[3];
    const float*     bih  = (const float*)d_in[4];
    const float*     bhh  = (const float*)d_in[5];
    const long long* idx  = (const long long*)d_in[6];
    float*           out  = (float*)d_out;

    const int Mrows = in_sizes[0] / D_S;

    // 1) S_next = ones (streaming nontemporal fill)
    const long long n = (long long)out_size;
    const int fb = (int)(((n + 3) / 4 + 255) / 256);
    fill_ones_kernel<<<fb, 256, 0, stream>>>(out, n);

    // 2) pack f16 weights into WMMA-B layout (workspace if available)
    const int use_ws = (ws_size >= (size_t)WS_BYTES) ? 1 : 0;
    _Float16* wpk = (_Float16*)d_ws;
    if (use_ws) {
        gru_pack_weights<<<(HALF_TOT + 255) / 256, 256, 0, stream>>>(Wih, Whh, wpk);
    }

    // 3) fused gather + dual-GEMM (WMMA f16->f32) + GRU gates + scatter
    const int gb = (Mrows + 127) / 128;
    gru_wmma_kernel<<<gb, 256, 0, stream>>>(msgs, S, Wih, Whh, bih, bhh,
                                            idx, wpk, out, Mrows, use_ws);
}
